// Hydra_25443386262245
// MI455X (gfx1250) — matrix operand order repaired and checked by reading, not compiled, not run on hardware
//
#include <hip/hip_runtime.h>
#include <hip/hip_bf16.h>

// ---------------------------------------------------------------------------
// Types for CDNA5 WMMA (wave32): v_wmma_f32_16x16x32_bf16
// ---------------------------------------------------------------------------
typedef __bf16 bf16_t;
typedef bf16_t bf16x16 __attribute__((ext_vector_type(16)));
typedef float  f32x8   __attribute__((ext_vector_type(8)));

union Frag {
    bf16x16      v;
    unsigned int d[8];   // 8 dwords = 16 bf16 (packed K-pairs)
};

__device__ __forceinline__ unsigned short f2bf(float f) {
    unsigned int u = __float_as_uint(f);
    unsigned int r = u + 0x7fffu + ((u >> 16) & 1u);   // RNE
    return (unsigned short)(r >> 16);
}

// ---------------------------------------------------------------------------
// CDNA5 async global->LDS path (ASYNCcnt), guarded so compilation never breaks
// Signature (probed via hipcc diagnostic): (int4 AS1*, int4 AS3*, int, int)
// ---------------------------------------------------------------------------
#if defined(__has_builtin)
#if __has_builtin(__builtin_amdgcn_global_load_async_to_lds_b128)
#define HAVE_ASYNC_LDS 1
#endif
#endif

typedef int v4i_t __attribute__((ext_vector_type(4)));
typedef __attribute__((address_space(1))) v4i_t* g_v4i_ptr;
typedef __attribute__((address_space(3))) v4i_t* l_v4i_ptr;

__device__ __forceinline__ void wait_async_lds() {
#ifdef HAVE_ASYNC_LDS
#if __has_builtin(__builtin_amdgcn_s_wait_asynccnt)
    __builtin_amdgcn_s_wait_asynccnt(0);
#else
    asm volatile("s_wait_asynccnt 0" ::: "memory");
#endif
#endif
}

// ---------------------------------------------------------------------------
// f32 -> bf16 convert (grid-stride)
// ---------------------------------------------------------------------------
__global__ __launch_bounds__(256) void cvt_f32_bf16(const float* __restrict__ src,
                                                    unsigned short* __restrict__ dst,
                                                    long long n) {
    long long i = (long long)blockIdx.x * blockDim.x + threadIdx.x;
    long long stride = (long long)gridDim.x * blockDim.x;
    for (; i < n; i += stride) dst[i] = f2bf(src[i]);
}

// ---------------------------------------------------------------------------
// LayerNorm (one block per row, D = 2048) -> bf16
// ---------------------------------------------------------------------------
__global__ __launch_bounds__(256) void ln_bf16_kernel(const float* __restrict__ X,
                                                      const float* __restrict__ w,
                                                      const float* __restrict__ b,
                                                      unsigned short* __restrict__ H,
                                                      int D) {
    const int row  = blockIdx.x;
    const int tid  = threadIdx.x;
    const int lane = tid & 31;
    const int wave = tid >> 5;
    const float* x = X + (size_t)row * D;

    float s = 0.f, s2 = 0.f;
    for (int i = tid; i < D; i += 256) {
        float v = x[i];
        s += v; s2 += v * v;
    }
    for (int off = 16; off; off >>= 1) {
        s  += __shfl_xor(s,  off, 32);
        s2 += __shfl_xor(s2, off, 32);
    }
    __shared__ float rs_[8], rs2_[8];
    if (lane == 0) { rs_[wave] = s; rs2_[wave] = s2; }
    __syncthreads();
    if (wave == 0) {
        s  = (lane < 8) ? rs_[lane]  : 0.f;
        s2 = (lane < 8) ? rs2_[lane] : 0.f;
        for (int off = 4; off; off >>= 1) {
            s  += __shfl_xor(s,  off, 32);
            s2 += __shfl_xor(s2, off, 32);
        }
        if (lane == 0) { rs_[0] = s; rs2_[0] = s2; }
    }
    __syncthreads();
    s = rs_[0]; s2 = rs2_[0];

    const float inv_d = 1.0f / (float)D;
    const float mu    = s * inv_d;
    const float var   = s2 * inv_d - mu * mu;
    const float rstd  = rsqrtf(var + 1e-5f);

    unsigned short* h = H + (size_t)row * D;
    for (int i = tid; i < D; i += 256) {
        float v = (x[i] - mu) * rstd * w[i] + b[i];
        h[i] = f2bf(v);
    }
}

// ---------------------------------------------------------------------------
// bf16 WMMA GEMM: C[MxN] = A[MxK] * B[KxN] (+bias[N]) (+resid[MxN])
// Block tile 128x128, BK = 32, 8 waves (4 along M x 2 along N),
// each wave owns 32x64 = 2x4 WMMA tiles -> 8 v_wmma per LDS stage.
// Double-buffered pipeline:
//   issue async A(t+1) + global B(t+1)  ->  8 WMMAs on tile t
//   ->  transpose-store B(t+1)  ->  s_wait_asynccnt 0  ->  barrier
// ---------------------------------------------------------------------------
template <bool HAS_BIAS, bool HAS_RESID>
__global__ __launch_bounds__(256) void gemm_bf16_wmma(const unsigned short* __restrict__ A,
                                                      const unsigned short* __restrict__ Bw,
                                                      float* __restrict__ C,
                                                      const float* __restrict__ bias,
                                                      const float* __restrict__ resid,
                                                      int M, int N, int K) {
    __shared__ __align__(16) unsigned short As[2][128][32];  // ping-pong, 16 KB
    __shared__ unsigned short Bs[2][128][34];                // ping-pong, 17 KB (padded)

    const int tid  = threadIdx.x;
    const int lane = tid & 31;
    const int wave = tid >> 5;
    const int wm   = wave & 3;   // 4 waves along M
    const int wn   = wave >> 2;  // 2 waves along N
    const int m    = lane & 15;
    const int half = lane >> 4;

    const int blockRow = blockIdx.y * 128;
    const int blockCol = blockIdx.x * 128;

    // cooperative-load coordinates
    const int arow = tid >> 1;           // 0..127
    const int acol = (tid & 1) * 16;     // 0 / 16
    const int brow = tid >> 3;           // 0..31  (k)
    const int bcol = (tid & 7) * 16;     // 0..112 (n)

    auto issueA = [&](int kt, int buf) {
        const unsigned short* ag = A + (size_t)(blockRow + arow) * K + kt + acol;
#ifdef HAVE_ASYNC_LDS
        __builtin_amdgcn_global_load_async_to_lds_b128(
            (g_v4i_ptr)ag, (l_v4i_ptr)&As[buf][arow][acol], 0, 0);
        __builtin_amdgcn_global_load_async_to_lds_b128(
            (g_v4i_ptr)(ag + 8), (l_v4i_ptr)&As[buf][arow][acol + 8], 0, 0);
#else
        uint4 a0 = *(const uint4*)ag;
        uint4 a1 = *(const uint4*)(ag + 8);
        unsigned int* asd = (unsigned int*)&As[buf][arow][acol];
        asd[0] = a0.x; asd[1] = a0.y; asd[2] = a0.z; asd[3] = a0.w;
        asd[4] = a1.x; asd[5] = a1.y; asd[6] = a1.z; asd[7] = a1.w;
#endif
    };
    auto loadB = [&](int kt, uint4& b0, uint4& b1) {
        const unsigned short* bg = Bw + (size_t)(kt + brow) * N + blockCol + bcol;
        b0 = *(const uint4*)bg;
        b1 = *(const uint4*)(bg + 8);
    };
    auto storeB = [&](int buf, uint4 b0, uint4 b1) {
        unsigned short be[16];
        *(uint4*)&be[0] = b0;
        *(uint4*)&be[8] = b1;
        #pragma unroll
        for (int e = 0; e < 16; ++e) Bs[buf][bcol + e][brow] = be[e];
    };

    f32x8 acc[2][4] = {};
    uint4 b0, b1;

    // ---- prologue: stage 0 ----
    issueA(0, 0);
    loadB(0, b0, b1);
    storeB(0, b0, b1);
    wait_async_lds();
    __syncthreads();

    for (int kt = 0; kt < K; kt += 32) {
        const int  cur  = (kt >> 5) & 1;
        const int  nxt  = cur ^ 1;
        const bool more = (kt + 32) < K;

        if (more) {
            issueA(kt + 32, nxt);        // async DMA overlaps the WMMAs below
            loadB(kt + 32, b0, b1);      // global loads in flight during compute
            if (kt + 64 < K) {           // global_prefetch_b8 two slabs ahead
                __builtin_prefetch(Bw + (size_t)(kt + 64 + brow) * N + blockCol + bcol, 0, 3);
            }
        }

        // --- assemble fragments per documented CDNA5 wave32 layouts ---
        Frag af[2], bfr[4];
        #pragma unroll
        for (int i = 0; i < 2; ++i) {
            const unsigned short* rowp = &As[cur][wm * 32 + i * 16 + m][0];
            #pragma unroll
            for (int v = 0; v < 8; ++v) {
                // A 16x32: vgpr v -> K = (v/4)*16 + half*8 + (v%4)*2 (pair)
                const int kb = ((v >> 2) << 4) + (half << 3) + ((v & 3) << 1);
                af[i].d[v] = *(const unsigned int*)&rowp[kb];
            }
        }
        #pragma unroll
        for (int j = 0; j < 4; ++j) {
            const unsigned short* rowp = &Bs[cur][wn * 64 + j * 16 + m][0];
            #pragma unroll
            for (int v = 0; v < 8; ++v) {
                // B 32x16: vgpr v -> K = (v/4)*16 + half*8 + (v%4)*2 (pair), N = lane%16
                const int kb = ((v >> 2) << 4) + (half << 3) + ((v & 3) << 1);
                bfr[j].d[v] = *(const unsigned int*)&rowp[kb];
            }
        }

        #pragma unroll
        for (int i = 0; i < 2; ++i) {
            #pragma unroll
            for (int j = 0; j < 4; ++j) {
                acc[i][j] = __builtin_amdgcn_wmma_f32_16x16x32_bf16(
                    false, af[i].v, false, bfr[j].v, (short)0, acc[i][j], false, false);
            }
        }

        if (more) {
            storeB(nxt, b0, b1);   // write next B tile (buffer was drained last iter)
            wait_async_lds();      // next A tile resident
            __syncthreads();       // make stage t+1 visible to all waves
        }
    }

    // Epilogue: C/D layout — vgpr r: row = r + half*8, col = lane%16
    #pragma unroll
    for (int j = 0; j < 4; ++j) {
        const int gcol = blockCol + wn * 64 + j * 16 + m;
        float bj = 0.f;
        if (HAS_BIAS) bj = bias[gcol];
        #pragma unroll
        for (int i = 0; i < 2; ++i) {
            #pragma unroll
            for (int r = 0; r < 8; ++r) {
                const int grow = blockRow + wm * 32 + i * 16 + half * 8 + r;
                float v = acc[i][j][r] + bj;
                if (HAS_RESID) v += resid[(size_t)grow * N + gcol];
                C[(size_t)grow * N + gcol] = v;
            }
        }
    }
    (void)M;
}

// ---------------------------------------------------------------------------
// RoPE (half rotation) applied in place to Q and K segments of qkv.
// One thread per (row, segment in {Q,K}, pair i in [0,512)).
// ---------------------------------------------------------------------------
__global__ __launch_bounds__(256) void rope_kernel(float* __restrict__ qkv,
                                                   int Mrows, int S) {
    const long long p = (long long)blockIdx.x * blockDim.x + threadIdx.x;
    if (p >= (long long)Mrows * 1024) return;
    const int row = (int)(p >> 10);
    const int t   = (int)(p & 1023);
    const int seg = t >> 9;        // 0 = Q, 1 = K
    const int i   = t & 511;       // pair index, rot_dim/2 = 512
    const int pos = row & (S - 1);

    float* q = qkv + (size_t)row * 6144 + seg * 2048 + 2 * i;
    // inv_freq = THETA^(-(2i)/rot_dim), rot_dim = 1024, THETA = 1e4
    const float invf = __expf(-9.210340371976184f * ((float)(2 * i) * (1.0f / 1024.0f)));
    const float ang  = (float)pos * invf;
    float c, s;
    __sincosf(ang, &s, &c);
    const float x1 = q[0], x2 = q[1];
    q[0] = x1 * c - x2 * s;
    q[1] = x2 * c + x1 * s;
}

// ---------------------------------------------------------------------------
// Per-row 1/||K_row|| (block per row)
// ---------------------------------------------------------------------------
__global__ __launch_bounds__(256) void knorm_kernel(const float* __restrict__ qkv,
                                                    float* __restrict__ rno, int D) {
    const int row  = blockIdx.x;
    const int tid  = threadIdx.x;
    const int lane = tid & 31;
    const int wave = tid >> 5;
    const float* k = qkv + (size_t)row * 6144 + 2048;

    float s = 0.f;
    for (int i = tid; i < D; i += 256) { float v = k[i]; s += v * v; }
    for (int off = 16; off; off >>= 1) s += __shfl_xor(s, off, 32);
    __shared__ float rs_[8];
    if (lane == 0) rs_[wave] = s;
    __syncthreads();
    if (wave == 0) {
        s = (lane < 8) ? rs_[lane] : 0.f;
        for (int off = 4; off; off >>= 1) s += __shfl_xor(s, off, 32);
        if (lane == 0) rno[row] = rsqrtf(s);
    }
}

// ---------------------------------------------------------------------------
// Deterministic chunked reduction: Apart[chunk][b][d] = sum over 256 rows of
// K_hat * V ;  then Avec[b][d] = sum over 16 chunks (fixed order).
// ---------------------------------------------------------------------------
__global__ __launch_bounds__(256) void apart_kernel(const float* __restrict__ qkv,
                                                    const float* __restrict__ rno,
                                                    float* __restrict__ Apart,
                                                    int S, int D) {
    const int idx   = blockIdx.x * blockDim.x + threadIdx.x;  // 16*4*2048
    const int chunk = idx >> 13;          // /8192
    const int rem   = idx & 8191;
    const int b     = rem >> 11;          // /2048
    const int d     = rem & 2047;
    const int rows_per_chunk = S >> 4;    // 256

    float acc = 0.f;
    const int row0 = b * S + chunk * rows_per_chunk;
    for (int s = 0; s < rows_per_chunk; ++s) {
        const size_t base = (size_t)(row0 + s) * 6144;
        acc += qkv[base + 2048 + d] * rno[row0 + s] * qkv[base + 4096 + d];
    }
    Apart[idx] = acc;
}

__global__ __launch_bounds__(256) void areduce_kernel(const float* __restrict__ Apart,
                                                      float* __restrict__ Avec) {
    const int i = blockIdx.x * blockDim.x + threadIdx.x;  // 4*2048
    float acc = 0.f;
    #pragma unroll
    for (int c = 0; c < 16; ++c) acc += Apart[c * 8192 + i];
    Avec[i] = acc;
}

// ---------------------------------------------------------------------------
// Z = (A broadcast over seq) ∘ Q  -> bf16
// ---------------------------------------------------------------------------
__global__ __launch_bounds__(256) void scaleq_kernel(const float* __restrict__ qkv,
                                                     const float* __restrict__ Avec,
                                                     unsigned short* __restrict__ Z,
                                                     int Mrows, int D, int S) {
    const long long idx = (long long)blockIdx.x * blockDim.x + threadIdx.x;
    if (idx >= (long long)Mrows * D) return;
    const int row = (int)(idx / D);
    const int d   = (int)(idx - (long long)row * D);
    const int b   = row / S;
    const float q = qkv[(size_t)row * 6144 + d];
    Z[idx] = f2bf(Avec[b * D + d] * q);
}

// ---------------------------------------------------------------------------
// launch
// ---------------------------------------------------------------------------
extern "C" void kernel_launch(void* const* d_in, const int* in_sizes, int n_in,
                              void* d_out, int out_size, void* d_ws, size_t ws_size,
                              hipStream_t stream) {
    const float* X     = (const float*)d_in[0];
    const float* ln_w  = (const float*)d_in[1];
    const float* ln_b  = (const float*)d_in[2];
    const float* W_in  = (const float*)d_in[3];
    const float* b_in  = (const float*)d_in[4];
    const float* W_out = (const float*)d_in[5];
    const float* b_out = (const float*)d_in[6];
    float* out = (float*)d_out;

    const int Bb = 4, S = 4096, D = 2048;
    const int Mrows = Bb * S;       // 16384
    const int N1 = 3 * D;           // 6144

    char* ws = (char*)d_ws;
    size_t off = 0;
    auto carve = [&](size_t bytes) -> char* {
        char* p = ws + off;
        off += (bytes + 255) & ~(size_t)255;
        return p;
    };
    unsigned short* Hbf    = (unsigned short*)carve((size_t)Mrows * D * 2);
    unsigned short* WinBf  = (unsigned short*)carve((size_t)D * N1 * 2);
    unsigned short* WoutBf = (unsigned short*)carve((size_t)D * D * 2);
    float*          qkv    = (float*)carve((size_t)Mrows * N1 * 4);
    float*          rno    = (float*)carve((size_t)Mrows * 4);
    float*          Apart  = (float*)carve((size_t)16 * Bb * D * 4);
    float*          Avec   = (float*)carve((size_t)Bb * D * 4);
    unsigned short* Zbf    = (unsigned short*)carve((size_t)Mrows * D * 2);
    (void)ws_size; (void)in_sizes; (void)n_in; (void)out_size;

    // 1. weight converts
    cvt_f32_bf16<<<4096, 256, 0, stream>>>(W_in, WinBf, (long long)D * N1);
    cvt_f32_bf16<<<2048, 256, 0, stream>>>(W_out, WoutBf, (long long)D * D);

    // 2. LayerNorm -> bf16
    ln_bf16_kernel<<<Mrows, 256, 0, stream>>>(X, ln_w, ln_b, Hbf, D);

    // 3. QKV GEMM (+b_in)
    gemm_bf16_wmma<true, false><<<dim3(N1 / 128, Mrows / 128), 256, 0, stream>>>(
        Hbf, WinBf, qkv, b_in, nullptr, Mrows, N1, D);

    // 4. RoPE on Q and K
    rope_kernel<<<(Mrows * 1024) / 256, 256, 0, stream>>>(qkv, Mrows, S);

    // 5. A = sum_s cos_sim(K) * V   (deterministic chunked reduce)
    knorm_kernel<<<Mrows, 256, 0, stream>>>(qkv, rno, D);
    apart_kernel<<<(16 * Bb * D) / 256, 256, 0, stream>>>(qkv, rno, Apart, S, D);
    areduce_kernel<<<(Bb * D) / 256, 256, 0, stream>>>(Apart, Avec);

    // 6. Z = A ∘ Q -> bf16
    scaleq_kernel<<<(int)(((long long)Mrows * D) / 256), 256, 0, stream>>>(
        qkv, Avec, Zbf, Mrows, D, S);

    // 7. Y = Z @ W_out + b_out + X
    gemm_bf16_wmma<true, true><<<dim3(D / 128, Mrows / 128), 256, 0, stream>>>(
        Zbf, WoutBf, out, b_out, X, Mrows, D, D);
}
